// Int8Linear_71957882077552
// MI455X (gfx1250) — 8-bit matrix operand order repaired and checked by reading, compiled, not run on hardware
//
#include <hip/hip_runtime.h>
#include <hip/hip_bf16.h>


typedef int v8i __attribute__((ext_vector_type(8)));

#define TILE_K     64
#define BLK_M      128
#define BLK_N      128
#define LDS_STRIDE 80

union FragA { v8i v; long long d[4]; };
union FragB { v8i v; int4 q[2]; };

template <int OFF>
__device__ __forceinline__ void async_load_b128(unsigned lds_addr, unsigned voff,
                                                const char* sbase) {
    asm volatile("global_load_async_to_lds_b128 %0, %1, %2 offset:%3"
                 :: "v"(lds_addr), "v"(voff), "s"(sbase), "n"(OFF) : "memory");
}

__device__ __forceinline__ void wait_async0() {
    asm volatile("s_wait_asynccnt 0x0" ::: "memory");
}

__device__ __forceinline__ unsigned lds_off32(const void* p) {
    return (unsigned)(unsigned long long)(uintptr_t)p;
}

__global__ __launch_bounds__(256) void quant_rows_kernel(
    const float* __restrict__ in, char* __restrict__ out8,
    float* __restrict__ scales, int K)
{
    const int row = blockIdx.x;
    const int t   = threadIdx.x;
    const float* rp = in + (size_t)row * K;
    const int chunks = K >> 10;

    float am = 0.0f;
    for (int c = 0; c < chunks; ++c) {
        float4 v = ((const float4*)rp)[c * 256 + t];
        am = fmaxf(am, fmaxf(fmaxf(fabsf(v.x), fabsf(v.y)),
                             fmaxf(fabsf(v.z), fabsf(v.w))));
    }

    __shared__ float red[256];
    red[t] = am;
    __syncthreads();
    for (int s = 128; s > 0; s >>= 1) {
        if (t < s) red[t] = fmaxf(red[t], red[t + s]);
        __syncthreads();
    }
    const float scale = fmaxf(red[0] * (1.0f / 127.0f), 1e-12f);
    const float inv   = 1.0f / scale;
    if (t == 0) scales[row] = scale;

    char* op = out8 + (size_t)row * K;
    for (int c = 0; c < chunks; ++c) {
        float4 v = ((const float4*)rp)[c * 256 + t];
        int a = (int)rintf(v.x * inv);
        int b = (int)rintf(v.y * inv);
        int g = (int)rintf(v.z * inv);
        int h = (int)rintf(v.w * inv);
        a = max(-128, min(127, a));
        b = max(-128, min(127, b));
        g = max(-128, min(127, g));
        h = max(-128, min(127, h));
        int packed = (a & 0xFF) | ((b & 0xFF) << 8) |
                     ((g & 0xFF) << 16) | ((h & 0xFF) << 24);
        ((int*)op)[c * 256 + t] = packed;
    }
}

__global__ __launch_bounds__(256) void int8_wmma_gemm_kernel(
    const char*  __restrict__ xq, const char* __restrict__ wq,
    const float* __restrict__ xscale, const float* __restrict__ wscale,
    const float* __restrict__ bias, float* __restrict__ out,
    int M, int N, int K)
{
    __shared__ alignas(16) char Xs[2][BLK_M * LDS_STRIDE];
    __shared__ alignas(16) char Ws[2][BLK_N * LDS_STRIDE];

    const int t     = threadIdx.x;
    const int lane  = t & 31;
    const int wv    = t >> 5;
    const int waveM = (wv & 1) * 64;
    const int waveN = (wv >> 1) * 32;
    const int bm    = blockIdx.y * BLK_M;
    const int bn    = blockIdx.x * BLK_N;

    v8i acc[4][2];
    for (int i = 0; i < 4; ++i)
        for (int j = 0; j < 2; ++j)
            for (int r = 0; r < 8; ++r)
                acc[i][j][r] = 0;

    const int ldrow = t >> 1;
    const int ldcol = (t & 1) * 32;
    const unsigned xvoff = (unsigned)((bm + ldrow) * (size_t)K + ldcol);
    const unsigned wvoff = (unsigned)((bn + ldrow) * (size_t)K + ldcol);
    const unsigned xlds[2] = {
        lds_off32(&Xs[0][ldrow * LDS_STRIDE + ldcol]),
        lds_off32(&Xs[1][ldrow * LDS_STRIDE + ldcol]) };
    const unsigned wlds[2] = {
        lds_off32(&Ws[0][ldrow * LDS_STRIDE + ldcol]),
        lds_off32(&Ws[1][ldrow * LDS_STRIDE + ldcol]) };

    const int nk  = K / TILE_K;
    const int l15 = lane & 15;
    const int lhi = lane >> 4;

    const char* bptr = Ws[0] + (waveN + l15) * LDS_STRIDE + lhi * 16;
    const char* aptr = Xs[0] + (waveM + l15) * LDS_STRIDE + lhi * 16;
    const int xbufstep = BLK_M * LDS_STRIDE;
    const int wbufstep = BLK_N * LDS_STRIDE;

    async_load_b128<0>(xlds[0], xvoff, xq);
    async_load_b128<16>(xlds[0], xvoff, xq);
    async_load_b128<0>(wlds[0], wvoff, wq);
    async_load_b128<16>(wlds[0], wvoff, wq);

    for (int kt = 0; kt < nk; ++kt) {
        const int buf = kt & 1;
        wait_async0();
        __syncthreads();

        if (kt + 1 < nk) {
            const char* xb = xq + (kt + 1) * TILE_K;
            const char* wb = wq + (kt + 1) * TILE_K;
            const int nb = buf ^ 1;
            async_load_b128<0>(xlds[nb], xvoff, xb);
            async_load_b128<16>(xlds[nb], xvoff, xb);
            async_load_b128<0>(wlds[nb], wvoff, wb);
            async_load_b128<16>(wlds[nb], wvoff, wb);
        }

        FragB fb[2];
        {
            const char* p = bptr + buf * wbufstep;
            fb[0].q[0] = *(const int4*)(p);
            fb[0].q[1] = *(const int4*)(p + 32);
            fb[1].q[0] = *(const int4*)(p + 16 * LDS_STRIDE);
            fb[1].q[1] = *(const int4*)(p + 16 * LDS_STRIDE + 32);
        }
        FragA fa[4];
        for (int i = 0; i < 4; ++i) {
            const char* p = aptr + buf * xbufstep + i * (16 * LDS_STRIDE);
            fa[i].d[0] = *(const long long*)(p);
            fa[i].d[1] = *(const long long*)(p + 8);
            fa[i].d[2] = *(const long long*)(p + 32);
            fa[i].d[3] = *(const long long*)(p + 40);
        }
        for (int i = 0; i < 4; ++i) {
            for (int j = 0; j < 2; ++j) {
                acc[i][j] = __builtin_amdgcn_wmma_i32_16x16x64_iu8(
                     true, fa[i].v,  true, fb[j].v,
                    acc[i][j],  false,  false);
            }
        }
    }

    const int mhi = (lane >> 4) * 8;
    for (int j = 0; j < 2; ++j) {
        const int col = bn + waveN + j * 16 + l15;
        const float wsc = wscale[col];
        const float bf  = __bfloat162float(__float2bfloat16(bias[col]));
        for (int i = 0; i < 4; ++i) {
            for (int r = 0; r < 8; ++r) {
                const int row = bm + waveM + i * 16 + mhi + r;
                float v = (float)acc[i][j][r] * xscale[row] * wsc;
                float vb  = __bfloat162float(__float2bfloat16(v));
                float res = __bfloat162float(__float2bfloat16(vb + bf));
                out[(size_t)row * N + col] = res;
            }
        }
    }
}

extern "C" void kernel_launch(void* const* d_in, const int* in_sizes, int n_in,
                              void* d_out, int out_size, void* d_ws, size_t ws_size,
                              hipStream_t stream)
{
    const float* x    = (const float*)d_in[0];
    const float* w    = (const float*)d_in[1];
    const float* bias = (const float*)d_in[2];
    float* out        = (float*)d_out;

    const int N = in_sizes[2];
    const int K = in_sizes[1] / N;
    const int M = in_sizes[0] / K;

    char*  wq     = (char*)d_ws;
    char*  xq     = wq + (size_t)N * K;
    float* wscale = (float*)(xq + (size_t)M * K);
    float* xscale = wscale + N;

    quant_rows_kernel<<<N, 256, 0, stream>>>(w, wq, wscale, K);
    quant_rows_kernel<<<M, 256, 0, stream>>>(x, xq, xscale, K);

    dim3 grid(N / BLK_N, M / BLK_M);
    int8_wmma_gemm_kernel<<<grid, 256, 0, stream>>>(
        xq, wq, xscale, wscale, bias, out, M, N, K);
}
